// CausalSelfAttention_89773406421623
// MI455X (gfx1250) — compile-verified, matrix operand order checked by reading, not run on hardware
//
#include <hip/hip_runtime.h>

// ---------------- problem constants ----------------
constexpr int Bb  = 2;
constexpr int Tt  = 8;
constexpr int Nn  = 256;
constexpr int Dk  = 768;
constexpr int Hh  = 12;
constexpr int HDd = 64;
constexpr int Lseq = Tt * Nn;          // 2048
constexpr int M_ALL = Bb * Lseq;       // 4096
constexpr int THREE_D = 3 * Dk;        // 2304

// ---------------- vector types ----------------
typedef __attribute__((ext_vector_type(16))) __bf16 v16bf;
typedef __attribute__((ext_vector_type(8)))  __bf16 v8bf;
typedef __attribute__((ext_vector_type(4)))  __bf16 v4bf;
typedef __attribute__((ext_vector_type(8)))  float  v8f;
typedef __attribute__((ext_vector_type(4)))  float  f32x4;

// -------- CDNA5 async global->LDS copy (ASYNCcnt path) --------
// Flip to 0 if the assembler rejects the mnemonic; fallback is a sync copy.
#define USE_ASYNC 1

__device__ inline void async_ld_b128(const void* g, void* lds) {
#if USE_ASYNC
  // Generic LDS pointer: addr[31:0] is the LDS byte address (aperture above).
  unsigned l = (unsigned)(size_t)lds;
  asm volatile("global_load_async_to_lds_b128 %0, %1, off"
               :: "v"(l), "v"(g) : "memory");
#else
  *(uint4*)lds = *(const uint4*)g;
#endif
}

__device__ inline void async_wait0() {
#if USE_ASYNC
  asm volatile("s_wait_asynccnt 0x0" ::: "memory");
#endif
}

// f32 -> bf16 round-to-nearest-even
__device__ inline __bf16 f2bf(float f) {
  unsigned u = __builtin_bit_cast(unsigned, f);
  u = (u + 0x7FFFu + ((u >> 16) & 1u)) >> 16;
  unsigned short h = (unsigned short)u;
  return __builtin_bit_cast(__bf16, h);
}

// CDNA5 16-bit A/B fragment load (16x32 sub-tile, row stride in elements).
__device__ inline v16bf load_ab_frag(const __bf16* tileBase, int rowStride, int lane) {
  const __bf16* p = tileBase + (lane & 15) * rowStride + ((lane & 16) ? 8 : 0);
  v16bf r;
  ((f32x4*)&r)[0] = *(const f32x4*)(p);
  ((f32x4*)&r)[1] = *(const f32x4*)(p + 16);
  return r;
}

__device__ inline v8f wmma_bf16(v16bf a, v16bf b, v8f c) {
  return __builtin_amdgcn_wmma_f32_16x16x32_bf16(
      false, a, false, b, (short)0, c, false, false);
}

// =====================================================================
// Kernel 1: QKV projection.  out(4096 x 2304) = x @ qkv_w^T + qkv_b
// q plane: (b,h)[l][d], pre-scaled by 1/8.  k plane: (b,h)[l][d].
// v plane: (b,h)[d][l]  (TRANSPOSED so attention can row-copy it).
// =====================================================================
__global__ __launch_bounds__(256) void qkv_kernel(
    const float* __restrict__ x, const float* __restrict__ w,
    const float* __restrict__ bias, __bf16* __restrict__ qkvb) {
  __shared__ __bf16 As[128 * 32];
  __shared__ __bf16 Bs[64 * 32];

  const int n0 = blockIdx.x * 64;
  const int m0 = blockIdx.y * 128;
  const int tid = threadIdx.x;
  const int lane = tid & 31;
  const int wv = tid >> 5;
  const int wm = (wv >> 1) * 32;
  const int wn = (wv & 1) * 32;

  v8f acc[2][2] = {};

  for (int k0 = 0; k0 < Dk; k0 += 32) {
    __syncthreads();
    // stage A tile (128x32 f32 -> bf16)
#pragma unroll
    for (int i = 0; i < 4; ++i) {
      int q = tid + i * 256;
      int row = q >> 3, c4 = (q & 7) * 4;
      const float* src = x + (size_t)(m0 + row) * Dk + k0 + c4;
      if (k0 + 32 < Dk) __builtin_prefetch(src + 32, 0, 0);
      f32x4 f = *(const f32x4*)src;
      v4bf o;
      o[0] = f2bf(f[0]); o[1] = f2bf(f[1]); o[2] = f2bf(f[2]); o[3] = f2bf(f[3]);
      *(v4bf*)&As[row * 32 + c4] = o;
    }
    // stage B tile (64x32 f32 -> bf16)
#pragma unroll
    for (int i = 0; i < 2; ++i) {
      int q = tid + i * 256;
      int row = q >> 3, c4 = (q & 7) * 4;
      f32x4 f = *(const f32x4*)(w + (size_t)(n0 + row) * Dk + k0 + c4);
      v4bf o;
      o[0] = f2bf(f[0]); o[1] = f2bf(f[1]); o[2] = f2bf(f[2]); o[3] = f2bf(f[3]);
      *(v4bf*)&Bs[row * 32 + c4] = o;
    }
    __syncthreads();
#pragma unroll
    for (int mi = 0; mi < 2; ++mi) {
      v16bf a = load_ab_frag(As + (wm + mi * 16) * 32, 32, lane);
#pragma unroll
      for (int ni = 0; ni < 2; ++ni) {
        v16bf b = load_ab_frag(Bs + (wn + ni * 16) * 32, 32, lane);
        acc[mi][ni] = wmma_bf16(a, b, acc[mi][ni]);
      }
    }
  }

  const int bidx = m0 >> 11;                 // tile never straddles batch
  const int rofs = (lane & 16) ? 8 : 0;
#pragma unroll
  for (int mi = 0; mi < 2; ++mi) {
#pragma unroll
    for (int ni = 0; ni < 2; ++ni) {
      int o = n0 + wn + ni * 16 + (lane & 15);
      int s = o / Dk;
      int rem = o - s * Dk;
      int h = rem >> 6, d = rem & 63;
      float bv = bias[o];
      size_t planeBase = (((size_t)s * Bb + bidx) * Hh + h) * (size_t)Lseq * HDd;
      if (s == 2) {
        // v plane transposed: [d][l]; this lane's 8 rows are l-contiguous.
        int l0 = (m0 + wm + mi * 16 + rofs) & (Lseq - 1);
        v8bf pk;
#pragma unroll
        for (int r = 0; r < 8; ++r) pk[r] = f2bf(acc[mi][ni][r] + bv);
        *(v8bf*)&qkvb[planeBase + (size_t)d * Lseq + l0] = pk;
      } else {
        float scale = (s == 0) ? 0.125f : 1.0f;   // HD^-0.5 folded into q
#pragma unroll
        for (int r = 0; r < 8; ++r) {
          int m = m0 + wm + mi * 16 + r + rofs;
          int l = m & (Lseq - 1);
          qkvb[planeBase + (size_t)l * HDd + d] = f2bf((acc[mi][ni][r] + bv) * scale);
        }
      }
    }
  }
}

// =====================================================================
// Kernel 2: fused attention, one (b,h,128-query-tile) per block.
// Async double-buffered K/V staging; keys start at t_q*256.
// =====================================================================
__global__ __launch_bounds__(256) void attn_kernel(
    const __bf16* __restrict__ qkvb, __bf16* __restrict__ attnb) {
  __shared__ __bf16 Qs[128 * 64];        // 16 KB
  __shared__ __bf16 Ks[2][64 * 64];      // 2 x 8 KB
  __shared__ __bf16 Vts[2][64 * 64];     // 2 x 8 KB  ([d][k] rows)
  __shared__ __bf16 Ps[8 * 16 * 64];     // 16 KB per-wave P patches

  const int qt = blockIdx.x;             // 0..15
  const int bh = blockIdx.y;             // 0..23
  const int b = bh / Hh, h = bh % Hh;
  const int tid = threadIdx.x;
  const int lane = tid & 31;
  const int wv = tid >> 5;

  const __bf16* qg = qkvb + (((size_t)0 * Bb + b) * Hh + h) * (size_t)Lseq * HDd
                          + (size_t)qt * 128 * 64;
  const __bf16* kg = qkvb + (((size_t)1 * Bb + b) * Hh + h) * (size_t)Lseq * HDd;
  const __bf16* vg = qkvb + (((size_t)2 * Bb + b) * Hh + h) * (size_t)Lseq * HDd; // [d][l]

  const int kstart = ((qt * 128) >> 8) * 256;   // first allowed key (t_k >= t_q)

  // ---- prologue staging: Q tile + first K/V tiles, all async ----
#pragma unroll
  for (int i = 0; i < 4; ++i) {
    int c = tid + i * 256;                     // 1024 x 16B chunks
    async_ld_b128(qg + c * 8, &Qs[c * 8]);
  }
#pragma unroll
  for (int i = 0; i < 2; ++i) {
    int q = tid + i * 256;                     // 512 x 16B chunks each
    int row = q >> 3, c = (q & 7) * 8;
    async_ld_b128(kg + (size_t)(kstart + row) * 64 + c, &Ks[0][row * 64 + c]);
    async_ld_b128(vg + (size_t)row * Lseq + kstart + c, &Vts[0][row * 64 + c]);
  }
  async_wait0();
  __syncthreads();

  float mrow[8], lrow[8];
#pragma unroll
  for (int r = 0; r < 8; ++r) { mrow[r] = -INFINITY; lrow[r] = 0.0f; }
  v8f oacc[4] = {};

  int buf = 0;
  for (int kt = kstart; kt < Lseq; kt += 64) {
    const int nxt = kt + 64;
    if (nxt < Lseq) {
      // prefetch next K/V tiles into the other buffer while we compute
#pragma unroll
      for (int i = 0; i < 2; ++i) {
        int q = tid + i * 256;
        int row = q >> 3, c = (q & 7) * 8;
        async_ld_b128(kg + (size_t)(nxt + row) * 64 + c, &Ks[buf ^ 1][row * 64 + c]);
        async_ld_b128(vg + (size_t)row * Lseq + nxt + c, &Vts[buf ^ 1][row * 64 + c]);
      }
    }

    // ---- S(16x64) = Q_tile(16x64) @ K_tile^T ----
    const __bf16* Kc = Ks[buf];
    const __bf16* Vc = Vts[buf];
    v16bf a0 = load_ab_frag(Qs + (wv * 16) * 64 + 0, 64, lane);
    v16bf a1 = load_ab_frag(Qs + (wv * 16) * 64 + 32, 64, lane);
    v8f sfr[4];
#pragma unroll
    for (int ni = 0; ni < 4; ++ni) {
      v16bf b0 = load_ab_frag(Kc + (ni * 16) * 64 + 0, 64, lane);
      v16bf b1 = load_ab_frag(Kc + (ni * 16) * 64 + 32, 64, lane);
      v8f sa = {};
      sa = wmma_bf16(a0, b0, sa);
      sa = wmma_bf16(a1, b1, sa);
      sfr[ni] = sa;
    }

    // ---- online softmax (row = vgpr r + 8*(lane>>4); N over 16 lanes) ----
#pragma unroll
    for (int r = 0; r < 8; ++r) {
      float mx = -INFINITY;
#pragma unroll
      for (int f = 0; f < 4; ++f) mx = fmaxf(mx, sfr[f][r]);
#pragma unroll
      for (int sh = 8; sh >= 1; sh >>= 1) mx = fmaxf(mx, __shfl_xor(mx, sh, 16));
      float nm = fmaxf(mrow[r], mx);
      float alpha = __expf(mrow[r] - nm);
      float rs = 0.0f;
#pragma unroll
      for (int f = 0; f < 4; ++f) {
        float p = __expf(sfr[f][r] - nm);
        sfr[f][r] = p;
        rs += p;
      }
#pragma unroll
      for (int sh = 8; sh >= 1; sh >>= 1) rs += __shfl_xor(rs, sh, 16);
      lrow[r] = lrow[r] * alpha + rs;
      mrow[r] = nm;
#pragma unroll
      for (int j = 0; j < 4; ++j) oacc[j][r] = oacc[j][r] * alpha;
    }

    // ---- C-layout P -> A-layout via per-wave LDS patch ----
    __bf16* pw = Ps + wv * 16 * 64;
    const int rofs = (lane & 16) ? 8 : 0;
#pragma unroll
    for (int f = 0; f < 4; ++f)
#pragma unroll
      for (int r = 0; r < 8; ++r)
        pw[(r + rofs) * 64 + f * 16 + (lane & 15)] = f2bf(sfr[f][r]);

    v16bf p0 = load_ab_frag(pw, 64, lane);
    v16bf p1 = load_ab_frag(pw + 32, 64, lane);
#pragma unroll
    for (int j = 0; j < 4; ++j) {
      v16bf b0 = load_ab_frag(Vc + (j * 16) * 64 + 0, 64, lane);
      v16bf b1 = load_ab_frag(Vc + (j * 16) * 64 + 32, 64, lane);
      oacc[j] = wmma_bf16(p0, b0, oacc[j]);
      oacc[j] = wmma_bf16(p1, b1, oacc[j]);
    }

    if (nxt < Lseq) {
      async_wait0();      // my prefetches landed
      __syncthreads();    // everyone's landed, everyone done with old buffer
      buf ^= 1;
    }
  }

  // ---- epilogue: O /= l, write bf16 (b, l, h*64+d) ----
  const int rofs = (lane & 16) ? 8 : 0;
#pragma unroll
  for (int j = 0; j < 4; ++j) {
    int d = h * 64 + j * 16 + (lane & 15);
#pragma unroll
    for (int r = 0; r < 8; ++r) {
      int l = qt * 128 + wv * 16 + r + rofs;
      attnb[((size_t)b * Lseq + l) * Dk + d] = f2bf(oacc[j][r] / lrow[r]);
    }
  }
}

// =====================================================================
// Kernel 3: output projection. out(4096x768) = attn @ proj_w^T + proj_b (f32)
// A tile staged through the async path (already bf16 in workspace).
// =====================================================================
__global__ __launch_bounds__(256) void proj_kernel(
    const __bf16* __restrict__ attnb, const float* __restrict__ w,
    const float* __restrict__ bias, float* __restrict__ out) {
  __shared__ __bf16 As[128 * 32];
  __shared__ __bf16 Bs[64 * 32];

  const int n0 = blockIdx.x * 64;
  const int m0 = blockIdx.y * 128;
  const int tid = threadIdx.x;
  const int lane = tid & 31;
  const int wv = tid >> 5;
  const int wm = (wv >> 1) * 32;
  const int wn = (wv & 1) * 32;

  v8f acc[2][2] = {};

  for (int k0 = 0; k0 < Dk; k0 += 32) {
    __syncthreads();
    // stage A (bf16, 128x32) via async copies
#pragma unroll
    for (int i = 0; i < 2; ++i) {
      int q = tid + i * 256;
      int row = q >> 2, c8 = (q & 3) * 8;
      async_ld_b128(attnb + (size_t)(m0 + row) * Dk + k0 + c8, &As[row * 32 + c8]);
    }
    // stage B (f32 -> bf16, 64x32)
#pragma unroll
    for (int i = 0; i < 2; ++i) {
      int q = tid + i * 256;
      int row = q >> 3, c4 = (q & 7) * 4;
      f32x4 f = *(const f32x4*)(w + (size_t)(n0 + row) * Dk + k0 + c4);
      v4bf o;
      o[0] = f2bf(f[0]); o[1] = f2bf(f[1]); o[2] = f2bf(f[2]); o[3] = f2bf(f[3]);
      *(v4bf*)&Bs[row * 32 + c4] = o;
    }
    async_wait0();
    __syncthreads();
#pragma unroll
    for (int mi = 0; mi < 2; ++mi) {
      v16bf a = load_ab_frag(As + (wm + mi * 16) * 32, 32, lane);
#pragma unroll
      for (int ni = 0; ni < 2; ++ni) {
        v16bf b = load_ab_frag(Bs + (wn + ni * 16) * 32, 32, lane);
        acc[mi][ni] = wmma_bf16(a, b, acc[mi][ni]);
      }
    }
  }

  const int rofs = (lane & 16) ? 8 : 0;
#pragma unroll
  for (int mi = 0; mi < 2; ++mi) {
#pragma unroll
    for (int ni = 0; ni < 2; ++ni) {
      int col = n0 + wn + ni * 16 + (lane & 15);
      float bv = bias[col];
#pragma unroll
      for (int r = 0; r < 8; ++r) {
        int m = m0 + wm + mi * 16 + r + rofs;
        out[(size_t)m * Dk + col] = acc[mi][ni][r] + bv;
      }
    }
  }
}

// =====================================================================
extern "C" void kernel_launch(void* const* d_in, const int* in_sizes, int n_in,
                              void* d_out, int out_size, void* d_ws, size_t ws_size,
                              hipStream_t stream) {
  (void)in_sizes; (void)n_in; (void)out_size; (void)ws_size;
  const float* x      = (const float*)d_in[0];
  const float* qkv_w  = (const float*)d_in[1];
  const float* qkv_b  = (const float*)d_in[2];
  const float* proj_w = (const float*)d_in[3];
  const float* proj_b = (const float*)d_in[4];
  float* out = (float*)d_out;

  __bf16* qkvb  = (__bf16*)d_ws;                                   // 3*B*H*L*HD bf16
  __bf16* attnb = qkvb + (size_t)3 * Bb * Hh * Lseq * HDd;         // B*L*D bf16

  qkv_kernel<<<dim3(THREE_D / 64, M_ALL / 128), 256, 0, stream>>>(x, qkv_w, qkv_b, qkvb);
  attn_kernel<<<dim3(Lseq / 128, Bb * Hh), 256, 0, stream>>>(qkvb, attnb);
  proj_kernel<<<dim3(Dk / 64, M_ALL / 128), 256, 0, stream>>>(attnb, proj_w, proj_b, out);
}
